// Generator_36945308680832
// MI455X (gfx1250) — matrix operand order repaired and checked by reading, not compiled, not run on hardware
//
#include <hip/hip_runtime.h>
#include <hip/hip_bf16.h>

typedef __attribute__((ext_vector_type(16))) _Float16 v16h;
typedef __attribute__((ext_vector_type(8)))  float    v8f;

union FragH { v16h h; uint4 u[2]; };
union HU   { _Float16 h; unsigned short u; };
struct Half4 { _Float16 v[4]; };

#define LRELU_SLOPE 0.2f
#define BN_EPS      1e-5f

// ---------------------------------------------------------------------------
// concat: xcat[i, 0:128] = normal_features[i], xcat[i, 128:192] = noise[i]
// ---------------------------------------------------------------------------
__global__ __launch_bounds__(256) void concat_kernel(const float* __restrict__ nf,
                                                     const float* __restrict__ nz,
                                                     float* __restrict__ xcat, int N) {
  int idx = blockIdx.x * blockDim.x + threadIdx.x;
  if (idx >= N * 48) return;
  int i = idx / 48, c = idx - i * 48;
  float4 v = (c < 32) ? ((const float4*)(nf + (size_t)i * 128))[c]
                      : ((const float4*)(nz + (size_t)i * 64))[c - 32];
  ((float4*)(xcat + (size_t)i * 192))[c] = v;
}

// ---------------------------------------------------------------------------
// edge scatter: agg[dst] += xcat[src]; deg[dst] += 1 (fp32 atomics, L2-resident:
// agg is 38.4 MB << 192 MB L2, so atomics resolve at L2 bandwidth)
// ---------------------------------------------------------------------------
__global__ __launch_bounds__(256) void scatter_kernel(const float* __restrict__ xcat,
                                                      const int* __restrict__ src,
                                                      const int* __restrict__ dst,
                                                      float* __restrict__ agg,
                                                      float* __restrict__ deg, int E) {
  int idx = blockIdx.x * blockDim.x + threadIdx.x;
  if (idx >= E * 48) return;
  int e = idx / 48, c = idx - e * 48;
  int s = src[e], d = dst[e];
  float4 v = ((const float4*)(xcat + (size_t)s * 192))[c];
  float* o = agg + (size_t)d * 192 + c * 4;
  unsafeAtomicAdd(o + 0, v.x);
  unsafeAtomicAdd(o + 1, v.y);
  unsafeAtomicAdd(o + 2, v.z);
  unsafeAtomicAdd(o + 3, v.w);
  if (c == 0) unsafeAtomicAdd(&deg[d], 1.0f);
}

// ---------------------------------------------------------------------------
// build A2 f16 [N, 384]:  cols 0..191 = agg/max(deg,1), cols 192..383 = xcat
// ---------------------------------------------------------------------------
__global__ __launch_bounds__(256) void build_a2_kernel(const float* __restrict__ agg,
                                                       const float* __restrict__ xcat,
                                                       const float* __restrict__ deg,
                                                       _Float16* __restrict__ A2, int N) {
  int idx = blockIdx.x * blockDim.x + threadIdx.x;
  if (idx >= N * 48) return;
  int i = idx / 48, c = idx - i * 48;
  float inv = 1.0f / fmaxf(deg[i], 1.0f);
  float4 a = ((const float4*)(agg + (size_t)i * 192))[c];
  float4 x = ((const float4*)(xcat + (size_t)i * 192))[c];
  Half4 ha = {{(_Float16)(a.x * inv), (_Float16)(a.y * inv),
               (_Float16)(a.z * inv), (_Float16)(a.w * inv)}};
  Half4 hx = {{(_Float16)x.x, (_Float16)x.y, (_Float16)x.z, (_Float16)x.w}};
  ((Half4*)(A2 + (size_t)i * 384))[c]       = ha;
  ((Half4*)(A2 + (size_t)i * 384 + 192))[c] = hx;
}

// ---------------------------------------------------------------------------
// pack fp32 [K,Nc] weights into per-lane WMMA B-fragment layout (f16 pairs):
// Bp[((nt*KT + kt)*32 + lane)*8 + v] packs B[k0][n] | B[k0+1][n]<<16,
// k0 = kt*32 + 16*(lane>>4) + 2*v, n = nt*16 + (lane&15)   (contiguous k order, matches A frag)
// split: rows >= split come from Wb (used to stack W_l over W_r).
// ---------------------------------------------------------------------------
__global__ __launch_bounds__(256) void packw_kernel(const float* __restrict__ Wa,
                                                    const float* __restrict__ Wb, int split,
                                                    int K, int Nc, unsigned* __restrict__ Bp) {
  int KT = K / 32, NT = Nc / 16;
  int idx = blockIdx.x * blockDim.x + threadIdx.x;
  if (idx >= KT * NT * 256) return;
  int v = idx & 7, lane = (idx >> 3) & 31, t = idx >> 8;
  int kt = t % KT, nt = t / KT;
  int hi = lane >> 4, lo = lane & 15;
  int k0 = kt * 32 + 16 * hi + 2 * v;
  int n  = nt * 16 + lo;
  float f0 = (k0 < split) ? Wa[(size_t)k0 * Nc + n] : Wb[(size_t)(k0 - split) * Nc + n];
  int k1 = k0 + 1;
  float f1 = (k1 < split) ? Wa[(size_t)k1 * Nc + n] : Wb[(size_t)(k1 - split) * Nc + n];
  HU h0, h1; h0.h = (_Float16)f0; h1.h = (_Float16)f1;
  Bp[idx] = (unsigned)h0.u | ((unsigned)h1.u << 16);
}

// ---------------------------------------------------------------------------
// Wave-level WMMA GEMM: writes columns [ntOff*16, (ntOff+NT)*16) of an output
// matrix with row stride ldOut. One wave owns a 16-row stripe; NT accumulator
// tiles stay in VGPRs (NT<=8 -> ~100 VGPRs, no spill at launch_bounds(256));
// A frag loaded once per k-tile (2x b128), B frags contiguous (2x b128/tile).
// EPI: 0 = bias+lrelu -> f16 ; 1 = bias+BN+lrelu -> f16 ; 2 = bias+lrelu -> f32
// ---------------------------------------------------------------------------
template <int KT, int NT, int EPI>
__global__ __launch_bounds__(256) void gemm_wmma(const _Float16* __restrict__ A,
                                                 const unsigned* __restrict__ Bp,
                                                 const float* __restrict__ bias,
                                                 const float* __restrict__ bn_g,
                                                 const float* __restrict__ bn_b,
                                                 const float* __restrict__ bn_m,
                                                 const float* __restrict__ bn_v,
                                                 _Float16* __restrict__ outH,
                                                 float* __restrict__ outF,
                                                 int M, int ntOff, int ldOut) {
  constexpr int K = KT * 32;
  const int lane   = threadIdx.x & 31;
  const int wave   = threadIdx.x >> 5;
  const int waveId = blockIdx.x * (blockDim.x >> 5) + wave;
  const int row0   = waveId * 16;
  if (row0 >= M) return;                 // wave-uniform exit; EXEC stays full
  const int hi = lane >> 4, lo = lane & 15;

  v8f acc[NT];
#pragma unroll
  for (int nt = 0; nt < NT; ++nt) acc[nt] = v8f{};

  const _Float16* arow = A + (size_t)(row0 + lo) * K;
  for (int kt = 0; kt < KT; ++kt) {
    // A fragment: contiguous k = 16*hi .. 16*hi+15 (matches packed B)
    FragH af;
    const uint4* ap = (const uint4*)(arow + kt * 32);
    af.u[0] = ap[2 * hi];
    af.u[1] = ap[2 * hi + 1];
#pragma unroll
    for (int nt = 0; nt < NT; ++nt) {
      const uint4* bp = (const uint4*)Bp + (size_t)((nt * KT + kt) * 32 + lane) * 2;
      FragH bf; bf.u[0] = bp[0]; bf.u[1] = bp[1];
      acc[nt] = __builtin_amdgcn_wmma_f32_16x16x32_f16(
          false, af.h, false, bf.h, (short)0, acc[nt], false, false);
    }
  }

#pragma unroll
  for (int nt = 0; nt < NT; ++nt) {
    const int n = (ntOff + nt) * 16 + lo;
    const float b = bias[n];
    float scale = 1.0f, shift = b;
    if (EPI == 1) {
      float inv = rsqrtf(bn_v[n] + BN_EPS);
      scale = bn_g[n] * inv;
      shift = (b - bn_m[n]) * scale + bn_b[n];
    }
#pragma unroll
    for (int r = 0; r < 8; ++r) {
      const int m = row0 + 8 * hi + r;   // C/D layout: VGPR r -> row r + 8*hi
      float val = acc[nt][r];
      val = (EPI == 1) ? (val * scale + shift) : (val + b);
      val = (val >= 0.0f) ? val : LRELU_SLOPE * val;
      if (EPI == 2) outF[(size_t)m * ldOut + n] = val;
      else          outH[(size_t)m * ldOut + n] = (_Float16)val;
    }
  }
}

// ---------------------------------------------------------------------------
extern "C" void kernel_launch(void* const* d_in, const int* in_sizes, int n_in,
                              void* d_out, int out_size, void* d_ws, size_t ws_size,
                              hipStream_t stream) {
  (void)n_in; (void)out_size; (void)ws_size;
  const int N = in_sizes[0] / 128;   // 50000
  const int E = in_sizes[2] / 2;     // 800000

  const float* nf   = (const float*)d_in[0];
  const float* nz   = (const float*)d_in[1];
  const int*   ei   = (const int*)d_in[2];
  const int*   src  = ei;
  const int*   dst  = ei + E;
  const float* W_l  = (const float*)d_in[4];
  const float* b_l  = (const float*)d_in[5];
  const float* W_r  = (const float*)d_in[6];
  const float* bn_g = (const float*)d_in[7];
  const float* bn_b = (const float*)d_in[8];
  const float* bn_m = (const float*)d_in[9];
  const float* bn_v = (const float*)d_in[10];
  const float* fc1w = (const float*)d_in[11];
  const float* fc1b = (const float*)d_in[12];
  const float* fc2w = (const float*)d_in[13];
  const float* fc2b = (const float*)d_in[14];
  const float* fc3w = (const float*)d_in[15];
  const float* fc3b = (const float*)d_in[16];
  const float* fc4w = (const float*)d_in[17];
  const float* fc4b = (const float*)d_in[18];

  char* ws = (char*)d_ws;
  size_t off = 0;
  auto take = [&](size_t bytes) { size_t r = off; off += (bytes + 255) & ~(size_t)255; return r; };
  float*     xcat = (float*)(ws + take((size_t)N * 192 * 4));
  float*     agg  = (float*)(ws + take((size_t)N * 192 * 4));
  float*     deg  = (float*)(ws + take((size_t)N * 4));
  _Float16*  A2   = (_Float16*)(ws + take((size_t)N * 384 * 2));
  _Float16*  h1   = (_Float16*)(ws + take((size_t)N * 256 * 2));
  _Float16*  h2   = (_Float16*)(ws + take((size_t)N * 128 * 2));
  _Float16*  h3   = (_Float16*)(ws + take((size_t)N * 64 * 2));
  _Float16*  h4   = (_Float16*)(ws + take((size_t)N * 64 * 2));
  unsigned*  Wp1  = (unsigned*)(ws + take((size_t)384 * 256 * 2));
  unsigned*  Wp2  = (unsigned*)(ws + take((size_t)256 * 128 * 2));
  unsigned*  Wp3  = (unsigned*)(ws + take((size_t)128 * 64 * 2));
  unsigned*  Wp4  = (unsigned*)(ws + take((size_t)64 * 64 * 2));
  unsigned*  Wp5  = (unsigned*)(ws + take((size_t)64 * 128 * 2));

  const int TB = 256;
  auto grid = [](long long n, int tb) { return (unsigned)((n + tb - 1) / tb); };

  hipMemsetAsync(agg, 0, (size_t)N * 192 * 4, stream);
  hipMemsetAsync(deg, 0, (size_t)N * 4, stream);

  concat_kernel<<<grid((long long)N * 48, TB), TB, 0, stream>>>(nf, nz, xcat, N);

  // weight packing (tiny; runs before GEMMs)
  packw_kernel<<<grid(12 * 16 * 256, TB), TB, 0, stream>>>(W_l, W_r, 192, 384, 256, Wp1);
  packw_kernel<<<grid( 8 *  8 * 256, TB), TB, 0, stream>>>(fc1w, fc1w, 256, 256, 128, Wp2);
  packw_kernel<<<grid( 4 *  4 * 256, TB), TB, 0, stream>>>(fc2w, fc2w, 128, 128,  64, Wp3);
  packw_kernel<<<grid( 2 *  4 * 256, TB), TB, 0, stream>>>(fc3w, fc3w,  64,  64,  64, Wp4);
  packw_kernel<<<grid( 2 *  8 * 256, TB), TB, 0, stream>>>(fc4w, fc4w,  64,  64, 128, Wp5);

  scatter_kernel<<<grid((long long)E * 48, TB), TB, 0, stream>>>(xcat, src, dst, agg, deg, E);

  build_a2_kernel<<<grid((long long)N * 48, TB), TB, 0, stream>>>(agg, xcat, deg, A2, N);

  const unsigned gblocks = grid((long long)(N / 16), 8);  // 8 waves (16 rows each) per block

  // layer 1: [N,384]@[384,256] + b_l, BN, lrelu -> h1 (f16); two column halves
  // (NT=8 each) to cap accumulator VGPRs at 64 and avoid scratch spills.
  gemm_wmma<12, 8, 1><<<gblocks, TB, 0, stream>>>(A2, Wp1,                b_l, bn_g, bn_b, bn_m, bn_v, h1, nullptr, N, 0, 256);
  gemm_wmma<12, 8, 1><<<gblocks, TB, 0, stream>>>(A2, Wp1 + 8 * 12 * 256, b_l, bn_g, bn_b, bn_m, bn_v, h1, nullptr, N, 8, 256);
  // fc1: [N,256]@[256,128] -> h2
  gemm_wmma< 8, 8, 0><<<gblocks, TB, 0, stream>>>(h1, Wp2, fc1b, nullptr, nullptr, nullptr, nullptr, h2, nullptr, N, 0, 128);
  // fc2: [N,128]@[128,64] -> h3
  gemm_wmma< 4, 4, 0><<<gblocks, TB, 0, stream>>>(h2, Wp3, fc2b, nullptr, nullptr, nullptr, nullptr, h3, nullptr, N, 0, 64);
  // fc3: [N,64]@[64,64] -> h4
  gemm_wmma< 2, 4, 0><<<gblocks, TB, 0, stream>>>(h3, Wp4, fc3b, nullptr, nullptr, nullptr, nullptr, h4, nullptr, N, 0, 64);
  // fc4: [N,64]@[64,128] -> d_out (f32)
  gemm_wmma< 2, 8, 2><<<gblocks, TB, 0, stream>>>(h4, Wp5, fc4b, nullptr, nullptr, nullptr, nullptr, nullptr, (float*)d_out, N, 0, 128);
}
